// TimeMix_17927193493904
// MI455X (gfx1250) — matrix operand order repaired and checked by reading, not compiled, not run on hardware
//
#include <hip/hip_runtime.h>
#include <hip/hip_bf16.h>
#include <stdint.h>

#define EPS 1e-8f

typedef __bf16 bf16_t;
typedef __attribute__((ext_vector_type(16))) __bf16 v16bf;
typedef __attribute__((ext_vector_type(8)))  float  v8f;

struct U4 { uint32_t x, y, z, w; };

__device__ __forceinline__ bf16_t f2bf(float f) {
    union { float f; uint32_t u; } a; a.f = f;
    uint32_t r = a.u + 0x7FFFu + ((a.u >> 16) & 1u);   // round to nearest even
    union { unsigned short s; bf16_t b; } o; o.s = (unsigned short)(r >> 16);
    return o.b;
}

// gfx1250 async global->LDS staging (tracked by ASYNCcnt)
__device__ __forceinline__ void async_load_b128(uint32_t lds_off, uint64_t gaddr) {
    asm volatile("global_load_async_to_lds_b128 %0, %1, off"
                 :: "v"(lds_off), "v"(gaddr) : "memory");
}
__device__ __forceinline__ void wait_async_4() {
    asm volatile("s_wait_asynccnt 0x4" ::: "memory");
}
__device__ __forceinline__ void wait_async_0() {
    asm volatile("s_wait_asynccnt 0x0" ::: "memory");
}

// ---------------------------------------------------------------------------
// fp32 -> bf16 conversion (weights)
// ---------------------------------------------------------------------------
__global__ void cvt_f32_bf16(const float* __restrict__ src,
                             bf16_t* __restrict__ dst, int n) {
    int i = blockIdx.x * blockDim.x + threadIdx.x;
    if (i < n) dst[i] = f2bf(src[i]);
}

// ---------------------------------------------------------------------------
// token-shift mix: xk/xv/xr = x*mix + x_prev*(1-mix), bf16 outputs
// ---------------------------------------------------------------------------
__global__ void mix_kernel(const float* __restrict__ x,
                           const float* __restrict__ mk,
                           const float* __restrict__ mv,
                           const float* __restrict__ mr,
                           bf16_t* __restrict__ xk,
                           bf16_t* __restrict__ xv,
                           bf16_t* __restrict__ xr,
                           int T, int D, int total) {
    int i = blockIdx.x * blockDim.x + threadIdx.x;
    if (i >= total) return;
    int d = i % D;
    int t = (i / D) % T;
    float xc = x[i];
    float xp = (t > 0) ? x[i - D] : 0.0f;
    float k = mk[d], v = mv[d], r = mr[d];
    xk[i] = f2bf(xc * k + xp * (1.0f - k));
    xv[i] = f2bf(xc * v + xp * (1.0f - v));
    xr[i] = f2bf(xc * r + xp * (1.0f - r));
}

// ---------------------------------------------------------------------------
// C(MxN) = act( A(MxK,bf16) * W(NxK,bf16)^T ), f32 out.  act: 0=none 1=sigmoid
// Block tile 128x128, 8 waves (wave32): 4(m) x 2(n), each wave 32x64.
// K stepped by 32 through double-buffered LDS filled with async-to-LDS DMA.
// Branch-free steady-state pipeline; last iteration peeled.
// ---------------------------------------------------------------------------
#define BM  128
#define BN  128
#define BK  32
#define LDT 40                    // padded LDS row stride in halves (80B)
#define TS  (BM * LDT)            // one tile region in halves (10240 B)

__global__ __launch_bounds__(256)
void gemm_bf16_wmma(const bf16_t* __restrict__ A,
                    const bf16_t* __restrict__ W,
                    float* __restrict__ C,
                    int M, int N, int K, int act) {
    __shared__ bf16_t smem[4 * TS];   // [buf0:A,B][buf1:A,B]

    const int tid  = threadIdx.x;
    const int lane = tid & 31;
    const int wave = tid >> 5;
    const int wm   = wave & 3;        // m offset 32*wm
    const int wn   = wave >> 2;       // n offset 64*wn
    const int mBase = blockIdx.y * BM;
    const int nBase = blockIdx.x * BN;

    const uint32_t smbase = (uint32_t)(uintptr_t)(&smem[0]);

    const v8f vzero = {0.f,0.f,0.f,0.f,0.f,0.f,0.f,0.f};
    v8f acc[2][4];
    #pragma unroll
    for (int i = 0; i < 2; ++i)
        #pragma unroll
        for (int j = 0; j < 4; ++j) acc[i][j] = vzero;

    // ISA lane patterns (16-bit A 16x32 / B 32x16 layouts)
    const int am   = lane & 15;
    const int aklo = (lane < 16) ? 0 : 8;    // halves {0..7,16..23} / {8..15,24..31}
    const int bn   = lane & 15;
    const int bk   = (lane < 16) ? 0 : 8;    // halves {0..7,16..23} / {8..15,24..31}

    // per-thread staging chunk coordinates (2 chunks for A, 2 for B)
    const int r0 = tid >> 2,          q0 = tid & 3;
    const int r1 = (tid + 256) >> 2,  q1 = tid & 3;   // rows 64..127

    union Frag { v16bf v; U4 q[2]; };

    const int nIter = K / BK;

    // per-thread invariant pieces of the staging addresses
    const bf16_t* gA0 = A + (size_t)(mBase + r0) * K + q0 * 8;
    const bf16_t* gA1 = A + (size_t)(mBase + r1) * K + q1 * 8;
    const bf16_t* gB0 = W + (size_t)(nBase + r0) * K + q0 * 8;
    const bf16_t* gB1 = W + (size_t)(nBase + r1) * K + q1 * 8;
    const uint32_t lo0 = (uint32_t)(r0 * LDT + q0 * 8) * 2;
    const uint32_t lo1 = (uint32_t)(r1 * LDT + q1 * 8) * 2;

    auto issue_tile = [&](int k0, int buf) {
        const uint32_t baseA = smbase + (uint32_t)(buf * 2 * TS) * 2;
        const uint32_t baseB = baseA + (uint32_t)TS * 2;
        async_load_b128(baseA + lo0, (uint64_t)(gA0 + k0));
        async_load_b128(baseA + lo1, (uint64_t)(gA1 + k0));
        async_load_b128(baseB + lo0, (uint64_t)(gB0 + k0));
        async_load_b128(baseB + lo1, (uint64_t)(gB1 + k0));
    };

    auto compute_tile = [&](int buf) {
        const bf16_t* As = &smem[buf * 2 * TS];
        const bf16_t* Bs = As + TS;
        Frag a[2], b[4];
        #pragma unroll
        for (int i = 0; i < 2; ++i) {
            int m = wm * 32 + i * 16 + am;
            a[i].q[0] = *(const U4*)(&As[m * LDT + aklo]);
            a[i].q[1] = *(const U4*)(&As[m * LDT + aklo + 16]);
        }
        #pragma unroll
        for (int j = 0; j < 4; ++j) {
            int n = wn * 64 + j * 16 + bn;
            b[j].q[0] = *(const U4*)(&Bs[n * LDT + bk]);
            b[j].q[1] = *(const U4*)(&Bs[n * LDT + bk + 16]);
        }
        #pragma unroll
        for (int i = 0; i < 2; ++i)
            #pragma unroll
            for (int j = 0; j < 4; ++j)
                acc[i][j] = __builtin_amdgcn_wmma_f32_16x16x32_bf16(
                    false, a[i].v, false, b[j].v, (short)0, acc[i][j], false, false);
    };

    issue_tile(0, 0);

    // steady state: branch-free (issue next, wait for current, compute current)
    for (int it = 0; it < nIter - 1; ++it) {
        issue_tile((it + 1) * BK, (it + 1) & 1);
        wait_async_4();               // 4 older ops (tile `it`) have landed
        __syncthreads();
        compute_tile(it & 1);
        __syncthreads();
    }
    // drain: last tile
    wait_async_0();
    __syncthreads();
    compute_tile((nIter - 1) & 1);

    // ---- epilogue: 16x16 f32 C layout (8 VGPRs; lanes>=16 hold M+8)
    const int rowAdd = (lane >> 4) * 8;
    const int colL   = lane & 15;
    #pragma unroll
    for (int i = 0; i < 2; ++i) {
        #pragma unroll
        for (int j = 0; j < 4; ++j) {
            #pragma unroll
            for (int r = 0; r < 8; ++r) {
                int row = mBase + wm * 32 + i * 16 + rowAdd + r;
                int col = nBase + wn * 64 + j * 16 + colL;
                float v = acc[i][j][r];
                if (act) v = 1.0f / (1.0f + __expf(-v));
                C[(size_t)row * N + col] = v;
            }
        }
    }
}

// ---------------------------------------------------------------------------
// sequential per-channel recurrence over T; emits bf16 for final GEMM
// ---------------------------------------------------------------------------
__global__ void recurrence_kernel(const float* __restrict__ k,
                                  const float* __restrict__ v,
                                  const float* __restrict__ r,
                                  const float* __restrict__ td,
                                  bf16_t* __restrict__ outb,
                                  int T, int D, int HS, int total) {
    int i = blockIdx.x * blockDim.x + threadIdx.x;   // i = b*D + d
    if (i >= total) return;
    int b = i / D, d = i % D;
    int h = d / HS;
    float wd = __expf(-__expf(td[h * HS]));          // time_decay[:,0]
    size_t base = (size_t)b * T * D + d;

    float kt = k[base], vt = v[base], rt = r[base];
    float prev = rt * kt * vt / (fabsf(kt) + EPS);
    outb[base] = f2bf(prev);
    for (int t = 1; t < T; ++t) {
        size_t o = base + (size_t)t * D;
        kt = k[o]; vt = v[o]; rt = r[o];
        prev = rt * (kt * vt + wd * prev) / (fabsf(kt + wd) + EPS);
        outb[o] = f2bf(prev);
    }
}

// ---------------------------------------------------------------------------
// s = concat([k_last*v_last + wd*(k_prev*v_prev), k_last], -1)  -> (B,H,2*HS)
// ---------------------------------------------------------------------------
__global__ void s_kernel(const float* __restrict__ k,
                         const float* __restrict__ v,
                         const float* __restrict__ td,
                         float* __restrict__ s_out,
                         int T, int D, int H, int HS, int total) {
    int i = blockIdx.x * blockDim.x + threadIdx.x;   // i = b*D + d
    if (i >= total) return;
    int b = i / D, d = i % D;
    int h = d / HS, hs = d % HS;
    float wd = __expf(-__expf(td[h * HS]));
    size_t o1 = ((size_t)b * T + (T - 1)) * D + d;
    size_t o2 = ((size_t)b * T + (T - 2)) * D + d;
    float k1 = k[o1], v1 = v[o1], k2 = k[o2], v2 = v[o2];
    float* sp = s_out + ((size_t)b * H + h) * (2 * HS);
    sp[hs]      = k1 * v1 + wd * (k2 * v2);
    sp[HS + hs] = k1;
}

// ---------------------------------------------------------------------------
extern "C" void kernel_launch(void* const* d_in, const int* in_sizes, int n_in,
                              void* d_out, int out_size, void* d_ws, size_t ws_size,
                              hipStream_t stream) {
    const int Bb = 4, T = 4096, D = 2048, H = 32, HS = 64;
    const int M = Bb * T;                 // 16384
    const int MD = M * D;                 // 33,554,432
    const int DD = D * D;                 // 4,194,304

    const float* x  = (const float*)d_in[0];
    const float* Wk = (const float*)d_in[1];
    const float* Wv = (const float*)d_in[2];
    const float* Wr = (const float*)d_in[3];
    const float* Wo = (const float*)d_in[4];
    const float* td = (const float*)d_in[5];
    const float* mk = (const float*)d_in[7];
    const float* mv = (const float*)d_in[8];
    const float* mr = (const float*)d_in[9];

    char* ws = (char*)d_ws;
    size_t off = 0;
    auto alloc = [&](size_t bytes) -> void* {
        void* p = ws + off; off += (bytes + 255) & ~(size_t)255; return p;
    };
    bf16_t* Wk_b = (bf16_t*)alloc((size_t)DD * 2);
    bf16_t* Wv_b = (bf16_t*)alloc((size_t)DD * 2);
    bf16_t* Wr_b = (bf16_t*)alloc((size_t)DD * 2);
    bf16_t* Wo_b = (bf16_t*)alloc((size_t)DD * 2);
    bf16_t* xk   = (bf16_t*)alloc((size_t)MD * 2);
    bf16_t* xv   = (bf16_t*)alloc((size_t)MD * 2);
    bf16_t* xr   = (bf16_t*)alloc((size_t)MD * 2);
    float*  kb   = (float*) alloc((size_t)MD * 4);
    float*  vb   = (float*) alloc((size_t)MD * 4);
    float*  rb   = (float*) alloc((size_t)MD * 4);
    bf16_t* outb = (bf16_t*)alloc((size_t)MD * 2);

    float* y = (float*)d_out;             // (B,T,D)
    float* s = y + (size_t)MD;            // (B,H,2*HS)

    // weights -> bf16
    {
        int blk = 256, grd = (DD + blk - 1) / blk;
        cvt_f32_bf16<<<grd, blk, 0, stream>>>(Wk, Wk_b, DD);
        cvt_f32_bf16<<<grd, blk, 0, stream>>>(Wv, Wv_b, DD);
        cvt_f32_bf16<<<grd, blk, 0, stream>>>(Wr, Wr_b, DD);
        cvt_f32_bf16<<<grd, blk, 0, stream>>>(Wo, Wo_b, DD);
    }
    // token-shift mix
    {
        int blk = 256, grd = (MD + blk - 1) / blk;
        mix_kernel<<<grd, blk, 0, stream>>>(x, mk, mv, mr, xk, xv, xr, T, D, MD);
    }
    // projections (WMMA bf16)
    dim3 ggrid(D / BN, M / BM);           // (16, 128)
    gemm_bf16_wmma<<<ggrid, 256, 0, stream>>>(xk, Wk_b, kb, M, D, D, 0);
    gemm_bf16_wmma<<<ggrid, 256, 0, stream>>>(xv, Wv_b, vb, M, D, D, 0);
    gemm_bf16_wmma<<<ggrid, 256, 0, stream>>>(xr, Wr_b, rb, M, D, D, 1);
    // recurrence
    {
        int total = Bb * D, blk = 256, grd = (total + blk - 1) / blk;
        recurrence_kernel<<<grd, blk, 0, stream>>>(kb, vb, rb, td, outb, T, D, HS, total);
    }
    // output projection -> y
    gemm_bf16_wmma<<<ggrid, 256, 0, stream>>>(outb, Wo_b, y, M, D, D, 0);
    // state s
    {
        int total = Bb * D, blk = 256, grd = (total + blk - 1) / blk;
        s_kernel<<<grd, blk, 0, stream>>>(kb, vb, td, s, T, D, H, HS, total);
    }
}
